// LLM_73804718014548
// MI455X (gfx1250) — matrix operand order repaired and checked by reading, not compiled, not run on hardware
//
#include <hip/hip_runtime.h>

// ---- model dims ----
#define LNUM 12
#define DDIM 768
#define NH   6
#define HDIM 128
#define FFD  3072
#define VSZ  65536
#define TSEQ 2048
#define EPSF 1e-5f
#define CAPF 30.0f

typedef __bf16 bf16_t;
typedef __attribute__((ext_vector_type(16))) __bf16 bf16x16;
typedef __attribute__((ext_vector_type(8)))  __bf16 bf16x8;
typedef __attribute__((ext_vector_type(4)))  __bf16 bf16x4;
typedef __attribute__((ext_vector_type(8)))  float  f32x8;
typedef long long i64;

// ---------- WMMA helper (CDNA5 16x16x32 bf16) ----------
__device__ __forceinline__ f32x8 wmma_bf16(bf16x16 a, bf16x16 b, f32x8 c) {
  return __builtin_amdgcn_wmma_f32_16x16x32_bf16(false, a, false, b, (short)0, c,
                                                 false, false);
}

// A fragment 16x32 (MxK): lane holds row m=lane%16; halves 0..7 -> k=8g..8g+7,
// halves 8..15 -> k=16+8g..23+8g  (g = lane/16).
__device__ __forceinline__ bf16x16 load_frag_a(const bf16_t* base, int ld) {
  const int lane = threadIdx.x & 31;
  const int m = lane & 15, g = lane >> 4;
  const bf16_t* p = base + m * ld + 8 * g;
  union { bf16x16 v; bf16x8 h[2]; } u;
  u.h[0] = *(const bf16x8*)(p);
  u.h[1] = *(const bf16x8*)(p + 16);
  return u.v;
}

// B fragment 32x16 (KxN) from row-major W[n][k]: lane holds col n=lane%16;
// halves 0..7 -> k=8g..8g+7, halves 8..15 -> k=16+8g..23+8g.
__device__ __forceinline__ bf16x16 load_frag_b(const bf16_t* base, int ld) {
  const int lane = threadIdx.x & 31;
  const int n = lane & 15, g = lane >> 4;
  const bf16_t* p = base + n * ld + 8 * g;
  union { bf16x16 v; bf16x8 h[2]; } u;
  u.h[0] = *(const bf16x8*)(p);
  u.h[1] = *(const bf16x8*)(p + 16);
  return u.v;
}

__device__ __forceinline__ bf16x4 cvt4(float4 v) {
  bf16x4 o;
  o[0] = (bf16_t)v.x; o[1] = (bf16_t)v.y; o[2] = (bf16_t)v.z; o[3] = (bf16_t)v.w;
  return o;
}

__device__ __forceinline__ float fast_tanh(float x) {
  const float e = __expf(2.f * x);
  return 1.f - 2.f / (e + 1.f);
}

// ---------- CDNA5 async global->LDS copy (ASYNCcnt-tracked) ----------
// 16B per lane: copies 16B from gptr into LDS at lds_off.  lds_off = wave-relative LDS byte addr
// (low 32 bits of a generic pointer to __shared__).
__device__ __forceinline__ void async_copy_b128(const bf16_t* lds_ptr,
                                                const bf16_t* gptr) {
  const unsigned lds_off = (unsigned)(unsigned long long)lds_ptr;
  asm volatile("global_load_async_to_lds_b128 %0, %1, off"
               :: "v"(lds_off), "v"(gptr) : "memory");
}
__device__ __forceinline__ void wait_async_le2() {
  asm volatile("s_wait_asynccnt 0x2" ::: "memory");
}
__device__ __forceinline__ void wait_async_0() {
  asm volatile("s_wait_asynccnt 0x0" ::: "memory");
}

// ---------- block reduce ----------
__device__ __forceinline__ float blockReduceSum(float v) {
  __shared__ float red[32];
  __syncthreads();
  for (int m = 16; m >= 1; m >>= 1) v += __shfl_xor(v, m, 32);
  const int lane = threadIdx.x & 31, w = threadIdx.x >> 5;
  const int nw = (blockDim.x + 31) >> 5;
  if (lane == 0) red[w] = v;
  __syncthreads();
  if (threadIdx.x == 0) {
    float s = 0.f;
    for (int i = 0; i < nw; ++i) s += red[i];
    red[0] = s;
  }
  __syncthreads();
  return red[0];
}

// ---------- embedding gather (float4) ----------
__global__ void k_embed(const int* __restrict__ idx, const float* __restrict__ emb,
                        float* __restrict__ x) {
  const int t = blockIdx.x;
  const float4* e = (const float4*)(emb + (i64)idx[t] * DDIM);
  float4* xr = (float4*)(x + (i64)t * DDIM);
  for (int i = threadIdx.x; i < DDIM / 4; i += blockDim.x) xr[i] = e[i];
}

// ---------- RMSNorm (f32 in, bf16 out: feeds WMMA GEMMs) ----------
__global__ __launch_bounds__(256) void k_rms(const float* __restrict__ x,
                                             const float* __restrict__ w,
                                             bf16_t* __restrict__ y, int n) {
  const int t = blockIdx.x;
  const float* xr = x + (i64)t * n;
  float ss = 0.f;
  for (int i = threadIdx.x; i < n; i += 256) { float v = xr[i]; ss += v * v; }
  ss = blockReduceSum(ss);
  const float r = rsqrtf(ss / (float)n + EPSF);
  bf16_t* yr = y + (i64)t * n;
  for (int i = threadIdx.x; i < n; i += 256) yr[i] = (bf16_t)(xr[i] * r * w[i]);
}

// ---------- f32 -> bf16 convert, vectorized ----------
__global__ void k_cvt_bf16(const float4* __restrict__ s, bf16x4* __restrict__ d,
                           i64 n4) {
  i64 i = (i64)blockIdx.x * blockDim.x + threadIdx.x;
  const i64 st = (i64)gridDim.x * blockDim.x;
  for (; i < n4; i += st) d[i] = cvt4(s[i]);
}

// ---------- WMMA GEMM, async double-buffered LDS pipeline ----------
//   C[M,N] = EP( A[M,K](bf16) * W[N,K]^T(bf16) )
#define BM 64
#define BN 64
#define BK 32
#define BKP 40  // padded LDS row (80B) -> conflict-free b128 fragment reads

template <int EP>
__global__ __launch_bounds__(256) void k_gemm(
    const bf16_t* __restrict__ A, const bf16_t* __restrict__ W,
    const float* __restrict__ R, float* __restrict__ Cf,
    bf16_t* __restrict__ Ch, int M, int N, int K) {
  __shared__ bf16_t As[2][BM][BKP];
  __shared__ bf16_t Bs[2][BN][BKP];
  const int tid = threadIdx.x;
  const int lane = tid & 31;
  const int wave = tid >> 5;        // 8 waves
  const int wn = wave & 3;          // 4 n-tiles
  const int wmp = wave >> 2;        // 2 m-pairs (two 16-row tiles each)
  const i64 n0 = (i64)blockIdx.x * BN;
  const i64 m0 = (i64)blockIdx.y * BM;

  // async tile copy: 256 threads x 16B covers one 64x32 bf16 tile
  const int row = tid >> 2;         // 0..63
  const int c8 = (tid & 3) * 8;     // half-offset 0,8,16,24
  const bf16_t* aG = A + (m0 + row) * (i64)K + c8;
  const bf16_t* bG = W + (n0 + row) * (i64)K + c8;

  f32x8 acc0 = {}, acc1 = {};
  const int nsteps = K / BK;

  // prologue: buffer 0
  async_copy_b128(&As[0][row][c8], aG);
  async_copy_b128(&Bs[0][row][c8], bG);

  for (int s = 0; s < nsteps; ++s) {
    const int cur = s & 1;
    if (s + 1 < nsteps) {  // prefetch next tile into the other buffer
      const int k1 = (s + 1) * BK;
      async_copy_b128(&As[cur ^ 1][row][c8], aG + k1);
      async_copy_b128(&Bs[cur ^ 1][row][c8], bG + k1);
      wait_async_le2();    // retire the 2 copies of the current buffer
    } else {
      wait_async_0();
    }
    __syncthreads();
    const bf16x16 bfr = load_frag_b(&Bs[cur][wn * 16][0], BKP);
    const bf16x16 a0 = load_frag_a(&As[cur][wmp * 32][0], BKP);
    const bf16x16 a1 = load_frag_a(&As[cur][wmp * 32 + 16][0], BKP);
    acc0 = wmma_bf16(a0, bfr, acc0);
    acc1 = wmma_bf16(a1, bfr, acc1);
    __syncthreads();       // all waves done reading before next overwrite
  }

  const int g = lane >> 4;
  const i64 nn = n0 + wn * 16 + (lane & 15);
#pragma unroll
  for (int sub = 0; sub < 2; ++sub) {
    const f32x8 acc = sub ? acc1 : acc0;
#pragma unroll
    for (int i = 0; i < 8; ++i) {
      const i64 mm = m0 + wmp * 32 + sub * 16 + i + 8 * g;
      const i64 o = mm * (i64)N + nn;
      const float v = acc[i];
      if (EP == 0)      Cf[o] = v;
      else if (EP == 1) Cf[o] = R[o] + v;
      else if (EP == 2) { const float rv = fmaxf(v, 0.f); Ch[o] = (bf16_t)(rv * rv); }
      else              Ch[o] = (bf16_t)(CAPF * fast_tanh(v * (1.0f / CAPF)));
    }
  }
}

// ---------- per-head q/k RMSNorm + RoPE + bf16 pack (v transposed) ----------
__global__ __launch_bounds__(128) void k_qk_rope(
    const float* __restrict__ qkv, const float* __restrict__ qw,
    const float* __restrict__ kw, bf16_t* __restrict__ qb,
    bf16_t* __restrict__ kb, bf16_t* __restrict__ vtb) {
  const int h = blockIdx.x, t = blockIdx.y, d = threadIdx.x;  // d in [0,128)
  const float* row = qkv + (i64)t * (3 * DDIM) + h * HDIM;
  const float q = row[d], k = row[DDIM + d], v = row[2 * DDIM + d];
  const float qs = blockReduceSum(q * q);
  const float ks = blockReduceSum(k * k);
  const float qn = q * rsqrtf(qs / (float)HDIM + EPSF) * qw[d];
  const float kn = k * rsqrtf(ks / (float)HDIM + EPSF) * kw[d];
  __shared__ float qsh[HDIM], ksh[HDIM];
  qsh[d] = qn; ksh[d] = kn;
  __syncthreads();
  const int p = (d < 64) ? d + 64 : d - 64;
  const float sgn = (d < 64) ? -1.f : 1.f;
  const float qr = sgn * qsh[p], kr = sgn * ksh[p];
  const int i2 = d & 63;
  const float inv = __powf(10000.f, -(2.f * (float)i2) / 128.f);
  const float ang = (float)t * inv;
  const float cc = __cosf(ang), ssn = __sinf(ang);
  qb[((i64)h * TSEQ + t) * HDIM + d] = (bf16_t)(qn * cc + qr * ssn);
  kb[((i64)h * TSEQ + t) * HDIM + d] = (bf16_t)(kn * cc + kr * ssn);
  vtb[((i64)(h * HDIM + d)) * TSEQ + t] = (bf16_t)v;
}

// ---------- flash attention: 4 waves/block, one wave = one 16-query tile ----------
// No __syncthreads: P round-trip is same-wave LDS RAW (DS ops in-order per wave;
// DScnt waits inserted by compiler). Waves have different trip counts.
__global__ __launch_bounds__(128) void k_attn(const bf16_t* __restrict__ qb,
                                              const bf16_t* __restrict__ kb,
                                              const bf16_t* __restrict__ vtb,
                                              bf16_t* __restrict__ o) {
  const int h = blockIdx.y;
  const int wave = threadIdx.x >> 5;
  const int t0 = blockIdx.x * 64 + wave * 16;
  const int lane = threadIdx.x & 31;
  const int g = lane >> 4, ln = lane & 15;
  __shared__ bf16_t Ps[4][16][40];
  bf16_t* myP = &Ps[wave][0][0];

  bf16x16 qf[4];
  const bf16_t* qbase = qb + ((i64)h * TSEQ + t0) * HDIM;
#pragma unroll
  for (int kc = 0; kc < 4; ++kc) qf[kc] = load_frag_a(qbase + kc * 32, HDIM);

  f32x8 O[8];
  float mrow[8], lrow[8];
#pragma unroll
  for (int i = 0; i < 8; ++i) { O[i] = {}; mrow[i] = -3.0e38f; lrow[i] = 0.f; }
  const float sc = 0.08838834764831845f;  // 1/sqrt(128)

  for (int s0 = 0; s0 <= t0 + 15; s0 += 32) {
    f32x8 S0 = {}, S1 = {};
#pragma unroll
    for (int kc = 0; kc < 4; ++kc) {
      const bf16x16 b0 = load_frag_b(kb + ((i64)h * TSEQ + s0) * HDIM + kc * 32, HDIM);
      const bf16x16 b1 =
          load_frag_b(kb + ((i64)h * TSEQ + s0 + 16) * HDIM + kc * 32, HDIM);
      S0 = wmma_bf16(qf[kc], b0, S0);
      S1 = wmma_bf16(qf[kc], b1, S1);
    }
#pragma unroll
    for (int i = 0; i < 8; ++i) {
      const int row = t0 + i + 8 * g;
      const int c0 = s0 + ln, c1 = s0 + 16 + ln;
      float v0 = (c0 <= row) ? S0[i] * sc : -1e30f;
      float v1 = (c1 <= row) ? S1[i] * sc : -1e30f;
      float rm = fmaxf(v0, v1);
      for (int m = 8; m >= 1; m >>= 1) rm = fmaxf(rm, __shfl_xor(rm, m, 32));
      const float mn = fmaxf(mrow[i], rm);
      const float scl = __expf(mrow[i] - mn);
      mrow[i] = mn;
      const float p0 = __expf(v0 - mn), p1 = __expf(v1 - mn);
      float rs = p0 + p1;
      for (int m = 8; m >= 1; m >>= 1) rs += __shfl_xor(rs, m, 32);
      lrow[i] = lrow[i] * scl + rs;
#pragma unroll
      for (int tl = 0; tl < 8; ++tl) O[tl][i] *= scl;
      myP[(i + 8 * g) * 40 + ln] = (bf16_t)p0;
      myP[(i + 8 * g) * 40 + 16 + ln] = (bf16_t)p1;
    }
    const bf16x16 pa = load_frag_a(myP, 40);
#pragma unroll
    for (int tl = 0; tl < 8; ++tl) {
      const bf16x16 vb =
          load_frag_b(vtb + ((i64)(h * HDIM + tl * 16)) * TSEQ + s0, TSEQ);
      O[tl] = wmma_bf16(pa, vb, O[tl]);
    }
  }
#pragma unroll
  for (int i = 0; i < 8; ++i) {
    const float inv = 1.f / lrow[i];
    const int row = t0 + i + 8 * g;
#pragma unroll
    for (int tl = 0; tl < 8; ++tl)
      o[(i64)row * DDIM + h * HDIM + tl * 16 + ln] = (bf16_t)(O[tl][i] * inv);
  }
}

// ---------- loss ----------
__global__ void k_zero(float* p) { if (threadIdx.x == 0 && blockIdx.x == 0) *p = 0.f; }

__global__ __launch_bounds__(256) void k_loss(const bf16_t* __restrict__ logits,
                                              const int* __restrict__ targets,
                                              float* __restrict__ acc) {
  const int t = blockIdx.x;
  const bf16_t* row = logits + (i64)t * VSZ;
  float m = -3.4e38f, s = 0.f;
  for (int i = threadIdx.x; i < VSZ; i += 256) {
    const float v = (float)row[i];
    if (v > m) { s = s * __expf(m - v) + 1.f; m = v; }
    else s += __expf(v - m);
  }
  __shared__ float ms[256], ssh[256];
  ms[threadIdx.x] = m; ssh[threadIdx.x] = s;
  __syncthreads();
  for (int st = 128; st >= 1; st >>= 1) {
    if (threadIdx.x < st) {
      const float m1 = ms[threadIdx.x], s1 = ssh[threadIdx.x];
      const float m2 = ms[threadIdx.x + st], s2 = ssh[threadIdx.x + st];
      const float mn = fmaxf(m1, m2);
      ssh[threadIdx.x] = s1 * __expf(m1 - mn) + s2 * __expf(m2 - mn);
      ms[threadIdx.x] = mn;
    }
    __syncthreads();
  }
  if (threadIdx.x == 0) {
    const float lt = (float)row[targets[t]];
    atomicAdd(acc, -(lt - ms[0] - __logf(ssh[0])));
  }
}

__global__ void k_loss_final(const float* acc, bf16_t* out) {
  if (threadIdx.x == 0 && blockIdx.x == 0)
    *out = (bf16_t)(*acc / (float)TSEQ);
}

// ---------- host ----------
extern "C" void kernel_launch(void* const* d_in, const int* in_sizes, int n_in,
                              void* d_out, int out_size, void* d_ws, size_t ws_size,
                              hipStream_t stream) {
  (void)in_sizes; (void)n_in; (void)ws_size;
  const int*   idx   = (const int*)d_in[0];
  const int*   tgt   = (const int*)d_in[1];
  const float* emb   = (const float*)d_in[2];
  const float* ln1   = (const float*)d_in[3];
  const float* qkvw  = (const float*)d_in[4];
  const float* qnw   = (const float*)d_in[5];
  const float* knw   = (const float*)d_in[6];
  const float* projw = (const float*)d_in[7];
  const float* ln2   = (const float*)d_in[8];
  const float* fc1w  = (const float*)d_in[9];
  const float* fc2w  = (const float*)d_in[10];
  const float* lnfw  = (const float*)d_in[11];
  const float* lmw   = (const float*)d_in[12];
  bf16_t* out = (bf16_t*)d_out;

  char* ws = (char*)d_ws;
  size_t off = 0;
  auto alloc = [&](size_t bytes) -> void* {
    void* p = ws + off;
    off = (off + bytes + 255) & ~(size_t)255;
    return p;
  };
  const size_t TD = (size_t)TSEQ * DDIM;
  float*  xA   = (float*)alloc(TD * 4);
  float*  xB   = (float*)alloc(TD * 4);
  float*  qkvb = (float*)alloc((size_t)TSEQ * 3 * DDIM * 4);
  bf16_t* hb   = (bf16_t*)alloc(TD * 2);
  bf16_t* ob   = (bf16_t*)alloc(TD * 2);
  bf16_t* ffb  = (bf16_t*)alloc((size_t)TSEQ * FFD * 2);
  bf16_t* qb   = (bf16_t*)alloc((size_t)NH * TSEQ * HDIM * 2 + 65536);
  bf16_t* kb   = (bf16_t*)alloc((size_t)NH * TSEQ * HDIM * 2 + 65536);
  bf16_t* vtb  = (bf16_t*)alloc((size_t)NH * TSEQ * HDIM * 2 + 65536);
  bf16_t* wqkv = (bf16_t*)alloc((size_t)3 * DDIM * DDIM * 2);
  bf16_t* wprj = (bf16_t*)alloc((size_t)DDIM * DDIM * 2);
  bf16_t* wf1  = (bf16_t*)alloc((size_t)FFD * DDIM * 2);
  bf16_t* wf2  = (bf16_t*)alloc((size_t)DDIM * FFD * 2);
  bf16_t* wlm  = (bf16_t*)alloc((size_t)VSZ * DDIM * 2);
  float* lossacc = (float*)alloc(256);

  auto cvt = [&](const float* s, bf16_t* d, size_t n) {
    k_cvt_bf16<<<1024, 256, 0, stream>>>((const float4*)s, (bf16x4*)d, (i64)(n / 4));
  };
  auto gemm = [&](const bf16_t* A, const bf16_t* W, const float* R, float* Cf,
                  bf16_t* Ch, int M, int N, int K, int ep) {
    dim3 grid(N / BN, M / BM);
    if (ep == 0)      k_gemm<0><<<grid, 256, 0, stream>>>(A, W, R, Cf, Ch, M, N, K);
    else if (ep == 1) k_gemm<1><<<grid, 256, 0, stream>>>(A, W, R, Cf, Ch, M, N, K);
    else if (ep == 2) k_gemm<2><<<grid, 256, 0, stream>>>(A, W, R, Cf, Ch, M, N, K);
    else              k_gemm<3><<<grid, 256, 0, stream>>>(A, W, R, Cf, Ch, M, N, K);
  };

  k_embed<<<TSEQ, 256, 0, stream>>>(idx, emb, xA);

  float* xcur = xA;
  float* xnxt = xB;
  for (int l = 0; l < LNUM; ++l) {
    cvt(qkvw  + (size_t)l * 3 * DDIM * DDIM, wqkv, (size_t)3 * DDIM * DDIM);
    cvt(projw + (size_t)l * DDIM * DDIM,     wprj, (size_t)DDIM * DDIM);
    cvt(fc1w  + (size_t)l * FFD * DDIM,      wf1,  (size_t)FFD * DDIM);
    cvt(fc2w  + (size_t)l * DDIM * FFD,      wf2,  (size_t)DDIM * FFD);

    k_rms<<<TSEQ, 256, 0, stream>>>(xcur, ln1 + (size_t)l * DDIM, hb, DDIM);
    gemm(hb, wqkv, nullptr, qkvb, nullptr, TSEQ, 3 * DDIM, DDIM, 0);
    k_qk_rope<<<dim3(NH, TSEQ), 128, 0, stream>>>(
        qkvb, qnw + (size_t)l * HDIM, knw + (size_t)l * HDIM, qb, kb, vtb);
    k_attn<<<dim3(TSEQ / 64, NH), 128, 0, stream>>>(qb, kb, vtb, ob);
    gemm(ob, wprj, xcur, xnxt, nullptr, TSEQ, DDIM, DDIM, 1);
    k_rms<<<TSEQ, 256, 0, stream>>>(xnxt, ln2 + (size_t)l * DDIM, hb, DDIM);
    gemm(hb, wf1, nullptr, nullptr, ffb, TSEQ, FFD, DDIM, 2);
    gemm(ffb, wf2, xnxt, xcur, nullptr, TSEQ, DDIM, FFD, 1);
  }

  k_rms<<<TSEQ, 256, 0, stream>>>(xcur, lnfw, hb, DDIM);
  cvt(lmw, wlm, (size_t)VSZ * DDIM);
  gemm(hb, wlm, nullptr, nullptr, out, TSEQ, VSZ, DDIM, 3);

  k_zero<<<1, 32, 0, stream>>>(lossacc);
  k_loss<<<TSEQ, 256, 0, stream>>>(out, tgt, lossacc);
  if ((i64)out_size > (i64)TSEQ * VSZ)
    k_loss_final<<<1, 32, 0, stream>>>(lossacc, out + (i64)TSEQ * VSZ);
}
